// SelfAttentionMechanism_V1_9259949490189
// MI455X (gfx1250) — matrix operand order repaired and checked by reading, not compiled, not run on hardware
//
#include <hip/hip_runtime.h>
#include <hip/hip_bf16.h>

typedef __attribute__((ext_vector_type(16))) _Float16 v16h;
typedef __attribute__((ext_vector_type(8)))  _Float16 v8h;
typedef __attribute__((ext_vector_type(4)))  _Float16 v4h;
typedef __attribute__((ext_vector_type(8)))  float    v8f;

#define DIM  1024
#define NSEQ 4096

// ---------------------------------------------------------------------------
// f32 -> f16 cast, 4 elements/thread, vectorized
// ---------------------------------------------------------------------------
__global__ __launch_bounds__(256)
void cast_f32_f16(const float* __restrict__ in, _Float16* __restrict__ out, int n) {
    int i = (blockIdx.x * blockDim.x + threadIdx.x) * 4;
    if (i < n) {
        float4 v = *(const float4*)(in + i);
        v4h h;
        h.x = (_Float16)v.x; h.y = (_Float16)v.y;
        h.z = (_Float16)v.z; h.w = (_Float16)v.w;
        *(v4h*)(out + i) = h;
    }
}

// ---------------------------------------------------------------------------
// Generic f16 GEMM, f32 accumulation via v_wmma_f32_16x16x32_f16.
// C[M,N] = alpha * A[M,K] @ B[K,N], A/B row-major f16.
// OUT_F32: store f32 row-major.  STORE_T: store f16 transposed (C^T, ld=M).
//
// Block: 256 threads = 8 wave32, tile 128x128, K-step 64 (2 WMMA k-slices).
// Waves arranged 4(M) x 2(N); each wave computes a 32x64 strip:
//   per K-step: 2 slices x (2 A-frags x 4 B-frags) = 16 WMMAs.
// Double-buffered LDS + register-staged global prefetch:
//   one barrier per K-step, global loads for tile k+1 in flight during
//   compute of tile k, commit to the alternate buffer after compute.
// ---------------------------------------------------------------------------
template<bool OUT_F32, bool STORE_T>
__global__ __launch_bounds__(256)
void gemm_f16_wmma(const _Float16* __restrict__ A,
                   const _Float16* __restrict__ B,
                   void* __restrict__ Cout,
                   int M, int N, int K, float alpha)
{
    constexpr int BM = 128, BN = 128, BK = 64;
    constexpr int AS = BK + 8;   // padded stride: 72 halves = 144 B = 36 dwords
    constexpr int BS = BK + 8;

    __shared__ _Float16 sA[2][BM * AS];   // [buf][row][k]
    __shared__ _Float16 sB[2][BN * BS];   // [buf][n][k]  (transposed)

    const int tid  = threadIdx.x;
    const int lane = tid & 31;
    const int wave = tid >> 5;
    const int ln15 = lane & 15;
    const int lhi  = lane >> 4;            // 0 or 1
    const int m0   = (wave & 3) * 32;      // wave row offset in tile
    const int n0   = (wave >> 2) * 64;     // wave col offset in tile

    const int blockN0 = blockIdx.x * BN;
    const int blockM0 = blockIdx.y * BM;

    // --- per-thread staging indices (constant across K loop) ---
    // A tile: 128 rows x 64 halves = 1024 v8h chunks; 4/thread (rows aR+32i)
    const int aR = tid >> 3;               // 0..31
    const int aK = (tid & 7) * 8;          // 0..56
    // B tile: 64 k-rows x 128 halves = 1024 v8h chunks; 4/thread (k = bKr+16i)
    const int bKr = tid >> 4;              // 0..15
    const int bN  = (tid & 15) * 8;        // 0..120

    const _Float16* gA = A + (size_t)(blockM0 + aR) * K + aK;
    const _Float16* gB = B + (size_t)bKr * N + blockN0 + bN;

    v8f acc[2][4] = {};
    v8h pa[4], pb[4];

    // --- prologue: stage tile 0, commit to buffer 0 ---
#pragma unroll
    for (int i = 0; i < 4; ++i) {
        pa[i] = *(const v8h*)(gA + (size_t)(32 * i) * K);
        pb[i] = *(const v8h*)(gB + (size_t)(16 * i) * N);
    }
#pragma unroll
    for (int i = 0; i < 4; ++i) {
        *(v8h*)(&sA[0][(aR + 32 * i) * AS + aK]) = pa[i];
#pragma unroll
        for (int h = 0; h < 8; ++h)
            sB[0][(bN + h) * BS + bKr + 16 * i] = pb[i][h];
    }
    __syncthreads();

    const int nIter = K / BK;
    for (int it = 0; it < nIter; ++it) {
        const int cur  = it & 1;
        const int nxt  = cur ^ 1;
        const bool more = (it + 1) < nIter;

        // --- issue global loads for next tile (overlap with compute) ---
        if (more) {
            const int kn = (it + 1) * BK;
#pragma unroll
            for (int i = 0; i < 4; ++i) {
                pa[i] = *(const v8h*)(gA + (size_t)(32 * i) * K + kn);
                pb[i] = *(const v8h*)(gB + (size_t)(16 * i + kn) * N);
            }
        }

        // --- compute 2 k-slices from current buffer: 16 WMMAs ---
        const _Float16* cA = sA[cur];
        const _Float16* cB = sB[cur];
#pragma unroll
        for (int ks = 0; ks < 2; ++ks) {
            v16h af[2], bf[4];
            const int akb = lhi * 8 + ks * 32;
            const int bkb = lhi * 8 + ks * 32;
#pragma unroll
            for (int im = 0; im < 2; ++im) {
                const int am = m0 + im * 16 + ln15;
                v8h alo = *(const v8h*)(cA + am * AS + akb);
                v8h ahi = *(const v8h*)(cA + am * AS + akb + 16);
                af[im] = __builtin_shufflevector(alo, ahi,
                    0,1,2,3,4,5,6,7,8,9,10,11,12,13,14,15);
            }
#pragma unroll
            for (int t = 0; t < 4; ++t) {
                const int bn = n0 + t * 16 + ln15;
                v8h blo = *(const v8h*)(cB + bn * BS + bkb);
                v8h bhi = *(const v8h*)(cB + bn * BS + bkb + 16);
                bf[t] = __builtin_shufflevector(blo, bhi,
                    0,1,2,3,4,5,6,7,8,9,10,11,12,13,14,15);
            }
#pragma unroll
            for (int im = 0; im < 2; ++im)
#pragma unroll
                for (int t = 0; t < 4; ++t)
                    acc[im][t] = __builtin_amdgcn_wmma_f32_16x16x32_f16(
                        false, af[im], false, bf[t], (short)0, acc[im][t],
                        false, false);
        }

        // --- commit staged registers to the alternate buffer ---
        if (more) {
#pragma unroll
            for (int i = 0; i < 4; ++i) {
                *(v8h*)(&sA[nxt][(aR + 32 * i) * AS + aK]) = pa[i];
#pragma unroll
                for (int h = 0; h < 8; ++h)
                    sB[nxt][(bN + h) * BS + bKr + 16 * i] = pb[i][h];
            }
        }
        __syncthreads();
    }

    // --- epilogue: lane holds C[m0+im*16+lhi*8+r][n0+t*16+ln15]
#pragma unroll
    for (int im = 0; im < 2; ++im) {
#pragma unroll
        for (int t = 0; t < 4; ++t) {
#pragma unroll
            for (int r = 0; r < 8; ++r) {
                int row = blockM0 + m0 + im * 16 + lhi * 8 + r;
                int col = blockN0 + n0 + t * 16 + ln15;
                float v = acc[im][t][r] * alpha;
                if (OUT_F32) {
                    ((float*)Cout)[(size_t)row * N + col] = v;
                } else if (STORE_T) {
                    ((_Float16*)Cout)[(size_t)col * M + row] = (_Float16)v;
                } else {
                    ((_Float16*)Cout)[(size_t)row * N + col] = (_Float16)v;
                }
            }
        }
    }
}

// ---------------------------------------------------------------------------
// In-place row softmax over f16 rows of length 4096. One block per row.
// ---------------------------------------------------------------------------
__global__ __launch_bounds__(256)
void softmax_rows_f16(_Float16* __restrict__ S, int Ncols)
{
    const int row = blockIdx.x;
    const int tid = threadIdx.x;
    _Float16* p = S + (size_t)row * Ncols;

    __shared__ float red[256];

    float v[16];
    v8h h0 = *(const v8h*)(p + tid * 16);
    v8h h1 = *(const v8h*)(p + tid * 16 + 8);
    float m = -3.4e38f;
#pragma unroll
    for (int i = 0; i < 8; ++i) { v[i] = (float)h0[i]; v[8 + i] = (float)h1[i]; }
#pragma unroll
    for (int i = 0; i < 16; ++i) m = fmaxf(m, v[i]);

    red[tid] = m;
    __syncthreads();
    for (int s = 128; s > 0; s >>= 1) {
        if (tid < s) red[tid] = fmaxf(red[tid], red[tid + s]);
        __syncthreads();
    }
    m = red[0];
    __syncthreads();

    float sum = 0.f;
#pragma unroll
    for (int i = 0; i < 16; ++i) { v[i] = __expf(v[i] - m); sum += v[i]; }

    red[tid] = sum;
    __syncthreads();
    for (int s = 128; s > 0; s >>= 1) {
        if (tid < s) red[tid] += red[tid + s];
        __syncthreads();
    }
    float inv = 1.0f / red[0];

    v8h o0, o1;
#pragma unroll
    for (int i = 0; i < 8; ++i) {
        o0[i] = (_Float16)(v[i] * inv);
        o1[i] = (_Float16)(v[8 + i] * inv);
    }
    *(v8h*)(p + tid * 16)     = o0;
    *(v8h*)(p + tid * 16 + 8) = o1;
}

// ---------------------------------------------------------------------------
// Orchestration: Q/K/V projections -> scores -> softmax -> output.
// ---------------------------------------------------------------------------
extern "C" void kernel_launch(void* const* d_in, const int* in_sizes, int n_in,
                              void* d_out, int out_size, void* d_ws, size_t ws_size,
                              hipStream_t stream)
{
    const float* y  = (const float*)d_in[0];
    const float* wq = (const float*)d_in[1];
    const float* wk = (const float*)d_in[2];
    const float* wv = (const float*)d_in[3];

    char* ws = (char*)d_ws;
    size_t off = 0;
    _Float16* Yh  = (_Float16*)(ws + off); off += (size_t)NSEQ * DIM * 2;  // 8 MB
    _Float16* Wqh = (_Float16*)(ws + off); off += (size_t)DIM * DIM * 2;   // 2 MB
    _Float16* Wkh = (_Float16*)(ws + off); off += (size_t)DIM * DIM * 2;
    _Float16* Wvh = (_Float16*)(ws + off); off += (size_t)DIM * DIM * 2;
    _Float16* Qh  = (_Float16*)(ws + off); off += (size_t)NSEQ * DIM * 2;  // 8 MB
    _Float16* Kth = (_Float16*)(ws + off); off += (size_t)NSEQ * DIM * 2;  // K^T [DIM x NSEQ]
    _Float16* Vh  = (_Float16*)(ws + off); off += (size_t)NSEQ * DIM * 2;
    _Float16* Sh  = (_Float16*)(ws + off); off += (size_t)NSEQ * NSEQ * 2; // 32 MB

    // casts to f16
    {
        int nY = NSEQ * DIM;
        cast_f32_f16<<<nY / 4 / 256, 256, 0, stream>>>(y, Yh, nY);
        int nW = DIM * DIM;
        cast_f32_f16<<<nW / 4 / 256, 256, 0, stream>>>(wq, Wqh, nW);
        cast_f32_f16<<<nW / 4 / 256, 256, 0, stream>>>(wk, Wkh, nW);
        cast_f32_f16<<<nW / 4 / 256, 256, 0, stream>>>(wv, Wvh, nW);
    }

    // Q = Y@Wq, K^T = (Y@Wk)^T, V = Y@Wv
    dim3 gProj(DIM / 128, NSEQ / 128);   // (8, 32)
    gemm_f16_wmma<false, false><<<gProj, 256, 0, stream>>>(Yh, Wqh, Qh,  NSEQ, DIM, DIM, 1.0f);
    gemm_f16_wmma<false, true ><<<gProj, 256, 0, stream>>>(Yh, Wkh, Kth, NSEQ, DIM, DIM, 1.0f);
    gemm_f16_wmma<false, false><<<gProj, 256, 0, stream>>>(Yh, Wvh, Vh,  NSEQ, DIM, DIM, 1.0f);

    // S = (Q @ K^T) / sqrt(d)   (scale folded into epilogue: 1/32)
    dim3 gS(NSEQ / 128, NSEQ / 128);     // (32, 32)
    gemm_f16_wmma<false, false><<<gS, 256, 0, stream>>>(Qh, Kth, Sh, NSEQ, NSEQ, DIM, 0.03125f);

    // softmax rows in place
    softmax_rows_f16<<<NSEQ, 256, 0, stream>>>(Sh, NSEQ);

    // O = P @ V  (f32 output)
    dim3 gO(DIM / 128, NSEQ / 128);
    gemm_f16_wmma<true, false><<<gO, 256, 0, stream>>>(Sh, Vh, d_out, NSEQ, DIM, NSEQ, 1.0f);
}
